// Decorrelation_80032420594485
// MI455X (gfx1250) — matrix operand order repaired and checked by reading, not compiled, not run on hardware
//
#include <hip/hip_runtime.h>

// ---- CDNA5 (gfx1250) wave32 WMMA types ----
typedef _Float16 f16;
typedef __attribute__((ext_vector_type(2)))  _Float16 f16x2;
typedef __attribute__((ext_vector_type(16))) _Float16 v16h;
typedef __attribute__((ext_vector_type(8)))  float    v8f;

#define DEG   9
#define KB    10            // Bernstein basis functions
#define DV    16            // variables
#define KKTOT 160           // DV*KB = GEMM contraction depth
#define TSTR  170           // LDS row stride in halves: 4B-aligned pairs, 85-dword (odd) stride -> conflict-free
#define WAVES 8
#define TPB   (WAVES * 32)
#define MAXBLOCKS 1024

__global__ __launch_bounds__(TPB)
void decorrelation_wmma_kernel(const float* __restrict__ x,
                               const float* __restrict__ params,
                               const float* __restrict__ prange,
                               float* __restrict__ out,
                               int Ntot, int numTiles)
{
    // B matrix (dense, transposed: [j][kk]) shared by all waves; A staging + x tile per wave.
    __shared__ f16   sP[DV * TSTR];
    __shared__ f16   sT[WAVES][16 * TSTR];
    __shared__ float sX[WAVES][256];
    __shared__ float sLo[DV], sInv[DV];

    const int tid  = threadIdx.x;
    const int lane = tid & 31;
    const int wave = tid >> 5;
    const int hl   = lane >> 4;    // which 16-lane half of the wave
    const int l15  = lane & 15;

    // ---- one-time block init: ranges + scatter 120 pair coeffs into dense f16 B ----
    if (tid < DV) {
        float lo = prange[tid], hi = prange[DV + tid];
        sLo[tid]  = lo;
        sInv[tid] = 1.0f / (hi - lo);
    }
    for (int idx = tid; idx < DV * KKTOT; idx += TPB) {
        int j  = idx / KKTOT;
        int kk = idx - j * KKTOT;
        int i  = kk / KB;
        int k  = kk - i * KB;
        float v = 0.0f;
        if (i < j) v = params[k * 120 + (j * (j - 1) / 2 + i)];  // pair (j,i), i<j
        sP[j * TSTR + kk] = (f16)v;
    }
    __syncthreads();

    // ---- hoist constant B fragments into registers (live across the tile loop) ----
    // B 32x16 f16 (same K order as A): lanes = cols, half h holds K 8h..8h+7 and 16+8h..16+8h+7
    v16h bfrag[5];
    #pragma unroll
    for (int c = 0; c < 5; ++c) {
        #pragma unroll
        for (int vg = 0; vg < 8; ++vg) {
            int koffB = (vg < 4) ? (hl * 8 + 2 * vg) : (16 + hl * 8 + 2 * (vg - 4));
            f16x2 pb = *(const f16x2*)&sP[l15 * TSTR + c * 32 + koffB];
            bfrag[c][2 * vg]     = pb.x;
            bfrag[c][2 * vg + 1] = pb.y;
        }
    }

    const float BIN[KB] = {1.f, 9.f, 36.f, 84.f, 126.f, 126.f, 84.f, 36.f, 9.f, 1.f};
    const int tileStep = gridDim.x * WAVES;
    const int mbase    = 8 * hl;

    for (int tile = blockIdx.x * WAVES + wave; tile < numTiles; tile += tileStep) {
        const int rowBase = tile * 16;
        const bool fullTile = (rowBase + 16 <= Ntot);   // wave-uniform

        // ---- coalesced 16x16 fp32 x-tile load into LDS ----
        if (fullTile) {
            const float4* src = (const float4*)(x + (long)rowBase * DV);
            float4* dst = (float4*)(&sX[wave][0]);
            dst[lane]      = src[lane];
            dst[lane + 32] = src[lane + 32];
        } else {
            for (int q = 0; q < 8; ++q) {
                int flat = q * 32 + lane;
                long g = (long)rowBase * DV + flat;
                sX[wave][flat] = (g < (long)Ntot * DV) ? x[g] : 0.0f;
            }
        }
        __asm__ volatile("" ::: "memory");  // same-wave LDS ops are in-order; block compile-time reordering

        // ---- A staging: t[row][i*10+k] = C(9,k) * u^k * (1-u)^(9-k) * x  (fp32 eval -> f16)
        //      x folded into the u-power chain: up[k] = x * u^k  -> 2 mults per basis value
        #pragma unroll
        for (int p = 0; p < 8; ++p) {
            int pr  = p * 32 + lane;
            int row = pr >> 4;
            int i   = pr & 15;
            float xv = sX[wave][row * DV + i];
            float u  = (xv - sLo[i]) * sInv[i];
            float w  = 1.0f - u;
            float up[KB], vp[KB];
            up[0] = xv; vp[0] = 1.0f;
            #pragma unroll
            for (int m = 1; m < KB; ++m) { up[m] = up[m - 1] * u; vp[m] = vp[m - 1] * w; }
            f16x2* dst = (f16x2*)&sT[wave][row * TSTR + i * KB];
            #pragma unroll
            for (int k = 0; k < KB; k += 2) {
                f16x2 pk;
                pk.x = (f16)(BIN[k]     * up[k]     * vp[DEG - k]);
                pk.y = (f16)(BIN[k + 1] * up[k + 1] * vp[DEG - k - 1]);
                dst[k >> 1] = pk;
            }
        }
        __asm__ volatile("" ::: "memory");

        // ---- prefetch all 5 A fragments, then run the WMMA chain with partial dscnt waits ----
        v16h afrag[5];
        #pragma unroll
        for (int c = 0; c < 5; ++c) {
            #pragma unroll
            for (int vg = 0; vg < 8; ++vg) {
                // A 16x32 f16 layout (ISA 7.12.2): lanes = rows, VGPR vg holds a K pair
                int koffA = (vg < 4) ? (hl * 8 + 2 * vg) : (16 + hl * 8 + 2 * (vg - 4));
                f16x2 pa = *(const f16x2*)&sT[wave][l15 * TSTR + c * 32 + koffA];
                afrag[c][2 * vg]     = pa.x;
                afrag[c][2 * vg + 1] = pa.y;
            }
        }
        v8f acc = {};
        #pragma unroll
        for (int c = 0; c < 5; ++c)
            acc = __builtin_amdgcn_wmma_f32_16x16x32_f16(false, afrag[c], false, bfrag[c],
                                                         (short)0, acc, false, false);

        // ---- epilogue: out = x + acc (D layout: VGPR v -> M = v + 8*hl, N = l15) ----
        float xo[8];
        #pragma unroll
        for (int v = 0; v < 8; ++v)
            xo[v] = sX[wave][(mbase + v) * DV + l15];

        if (fullTile) {
            // unconditional stores: one base pointer, immediate row offsets (64B apart)
            float* po = out + (long)(rowBase + mbase) * DV + l15;
            #pragma unroll
            for (int v = 0; v < 8; ++v)
                po[v * DV] = xo[v] + acc[v];
        } else {
            #pragma unroll
            for (int v = 0; v < 8; ++v) {
                int grow = rowBase + mbase + v;
                if (grow < Ntot)
                    out[(long)grow * DV + l15] = xo[v] + acc[v];
            }
        }
        __asm__ volatile("" ::: "memory");  // keep sX/sT reuse ordered across loop iterations
    }
}

extern "C" void kernel_launch(void* const* d_in, const int* in_sizes, int n_in,
                              void* d_out, int out_size, void* d_ws, size_t ws_size,
                              hipStream_t stream) {
    const float* x      = (const float*)d_in[0];   // [N,16] f32
    const float* params = (const float*)d_in[1];   // [10,120] f32
    const float* pr     = (const float*)d_in[2];   // [2,16] f32
    float* out          = (float*)d_out;           // [N,16] f32

    int Ntot     = in_sizes[0] / DV;
    int numTiles = (Ntot + 15) / 16;
    int blocks   = (numTiles + WAVES - 1) / WAVES;
    if (blocks > MAXBLOCKS) blocks = MAXBLOCKS;

    decorrelation_wmma_kernel<<<blocks, TPB, 0, stream>>>(x, params, pr, out, Ntot, numTiles);
}
